// DecoderOnlyAttention_66228395705057
// MI455X (gfx1250) — matrix operand order repaired and checked by reading, not compiled, not run on hardware
//
#include <hip/hip_runtime.h>
#include <hip/hip_bf16.h>

// ---------------------------------------------------------------------------
// DecoderOnlyAttention for MI455X (gfx1250): bf16 WMMA pipeline, fp32 accum,
// TDM (tensor_load_to_lds) staging for row-major tiles.
// B=2, S=2048, D=1024, H=16, hd=64.
// ---------------------------------------------------------------------------

typedef __attribute__((ext_vector_type(16))) __bf16 v16bf;
typedef __attribute__((ext_vector_type(8)))  float  v8f;

#define B_SZ 2
#define S_SZ 2048
#define D_SZ 1024
#define H_SZ 16
#define HD_SZ 64

#if defined(__gfx1250__) && __has_builtin(__builtin_amdgcn_tensor_load_to_lds)
#define HAVE_TDM 1
#endif

#ifdef HAVE_TDM
typedef __attribute__((ext_vector_type(4))) unsigned int v4u;
typedef __attribute__((ext_vector_type(8))) int          v8i;
typedef __attribute__((ext_vector_type(4))) int          v4i;

// Issue a 2-D TDM load of a [tile_d1 rows x tile_d0 elems] bf16 tile into LDS.
// Tensor is row-major [tensor_d1][tensor_d0] with row stride t0_stride elems.
// D# layout per CDNA5 ISA (group0 128b, group1 256b).
// This toolchain exposes the 6-arg builtin:
//   (uint32x4 g0, int32x8 g1, int32x4 g2, int32x4 g3, int32x8 pad, i32 cpol)
__device__ __forceinline__ void tdm_load_2d_bf16(unsigned int lds_off,
                                                 const void* gaddr,
                                                 unsigned int tensor_d0,
                                                 unsigned int tensor_d1,
                                                 unsigned int t0_stride,
                                                 unsigned int tile_d0,
                                                 unsigned int tile_d1) {
    unsigned long long ga = (unsigned long long)gaddr;
    v4u g0;
    g0[0] = 1u;                                        // count=1, user desc
    g0[1] = lds_off;                                   // LDS byte address
    g0[2] = (unsigned int)(ga & 0xffffffffu);          // global_addr[31:0]
    g0[3] = (unsigned int)((ga >> 32) & 0x1ffffffu)    // global_addr[56:32]
            | (2u << 30);                              // type = 2 ("image")
    v8i g1;
    g1[0] = (int)(1u << 16);                           // data_size=1 (2 bytes)
    g1[1] = (int)((tensor_d0 & 0xffffu) << 16);        // tensor_dim0[15:0]
    g1[2] = (int)((tensor_d0 >> 16) | ((tensor_d1 & 0xffffu) << 16));
    g1[3] = (int)((tensor_d1 >> 16) | (tile_d0 << 16));// tile_dim0
    g1[4] = (int)(tile_d1 & 0xffffu);                  // tile_dim1, tile_dim2=0
    g1[5] = (int)t0_stride;                            // dim0 stride[31:0]
    g1[6] = 0;                                         // stride[47:32], dim1 stride lo
    g1[7] = 0;
    v4i z4 = {0, 0, 0, 0};                             // 2-D: groups 2/3 unused
    v8i z8 = {0, 0, 0, 0, 0, 0, 0, 0};
    __builtin_amdgcn_tensor_load_to_lds(g0, g1, z4, z4, z8, 0);
}
#endif

__device__ __forceinline__ unsigned short f2bf(float x) {
    unsigned int u = __float_as_uint(x);
    unsigned int r = (u + 0x7fffu + ((u >> 16) & 1u)) >> 16;
    return (unsigned short)r;
}

// A-fragment, 16x32 bf16 (MxK), row-major source with leading dim ld (elems).
__device__ __forceinline__ v16bf load_a_frag(const unsigned short* p, int ld,
                                             int m_base, int k_base, int lane) {
    const int m = m_base + (lane & 15);
    const int half = lane >> 4;
    const unsigned int* row = (const unsigned int*)(p + m * ld + k_base);
    union { v16bf v; unsigned int u[8]; } f;
#pragma unroll
    for (int i = 0; i < 4; ++i) f.u[i]     = row[half * 8 + i];
#pragma unroll
    for (int i = 0; i < 4; ++i) f.u[4 + i] = row[half * 8 + 4 + i];
    return f.v;
}

// B-fragment, 32x16 bf16 (KxN), from K-transposed storage pT[N][K], ldk elems.
__device__ __forceinline__ v16bf load_b_fragT(const unsigned short* pT, int ldk,
                                              int n_base, int k_base, int lane) {
    const int n = n_base + (lane & 15);
    const int half = lane >> 4;
    const unsigned int* col = (const unsigned int*)(pT + n * ldk + k_base + half * 16);
    union { v16bf v; unsigned int u[8]; } f;
#pragma unroll
    for (int i = 0; i < 8; ++i) f.u[i] = col[i];
    return f.v;
}

// ---------------------------------------------------------------------------
// Kernel 1: fp32 -> bf16 conversion
// ---------------------------------------------------------------------------
__global__ __launch_bounds__(256) void cvt_bf16_kernel(const float* __restrict__ src,
                                                       unsigned short* __restrict__ dst,
                                                       int n) {
    int i = blockIdx.x * 256 + threadIdx.x;
    if (i < n) dst[i] = f2bf(src[i]);
}

// ---------------------------------------------------------------------------
// Shared GEMM body: C[128 x 128] tile per 256-thread block (8 waves), K-step 32.
// Each wave owns a 16-row strip and 8 WMMA accumulators across N.
// A tile staged by TDM (row-major); B tile staged transposed by all threads.
// ---------------------------------------------------------------------------
template <int LDW>  // leading dim of weight matrix (N total)
__device__ __forceinline__ void gemm_tile_128x128(
        const unsigned short* __restrict__ A16, int lda, int m_rows,
        const unsigned short* __restrict__ W16,
        int m0, int n0, int tid, v8f (&acc)[8],
        unsigned short* As /*128*32*/, unsigned short* BsT /*128*32*/) {
    const int lane = tid & 31;
    const int wv = tid >> 5;

    for (int k0 = 0; k0 < D_SZ; k0 += 32) {
        __syncthreads();
#ifdef HAVE_TDM
        if (wv == 0) {
            tdm_load_2d_bf16((unsigned int)(size_t)(void*)As,
                             (const void*)(A16 + (size_t)m0 * lda + k0),
                             (unsigned int)lda, (unsigned int)m_rows,
                             (unsigned int)lda, 32u, 128u);
        }
#else
        {
            unsigned int* dst = (unsigned int*)As;
            const unsigned int* src = (const unsigned int*)A16;
#pragma unroll
            for (int rep = 0; rep < 8; ++rep) {
                int i = tid + 256 * rep;          // [128 rows][16 dwords]
                int row = i >> 4, colw = i & 15;
                dst[i] = src[(m0 + row) * (lda / 2) + (k0 >> 1) + colw];
            }
        }
#endif
        // Stage B tile transposed: W[k0..+32][n0..+128] -> BsT[n][k].
#pragma unroll
        for (int rep = 0; rep < 16; ++rep) {
            int i = tid + 256 * rep;              // over [32][128]
            int kk = i >> 7, nn = i & 127;
            BsT[nn * 32 + kk] = W16[(k0 + kk) * LDW + n0 + nn];
        }
        if (k0 + 32 < D_SZ) {   // prefetch next W tile chunk
            __builtin_prefetch(&W16[(k0 + 32 + (tid >> 3)) * LDW + n0 + ((tid & 7) << 4)], 0, 0);
        }
#ifdef HAVE_TDM
        if (wv == 0) __builtin_amdgcn_s_wait_tensorcnt(0);
#endif
        __syncthreads();

        v16bf a = load_a_frag(As, 32, wv * 16, 0, lane);
#pragma unroll
        for (int j = 0; j < 8; ++j) {
            v16bf b = load_b_fragT(BsT, 32, j * 16, 0, lane);
            acc[j] = __builtin_amdgcn_wmma_f32_16x16x32_bf16(
                false, a, false, b, (short)0, acc[j], false, false);
        }
    }
}

// ---------------------------------------------------------------------------
// Kernel 2: QKV GEMM, scatter to Q/K/V buffers in [B,H,S,64] bf16 layout.
// ---------------------------------------------------------------------------
__global__ __launch_bounds__(256) void qkv_gemm_kernel(
        const unsigned short* __restrict__ hs16,
        const unsigned short* __restrict__ Wa16,
        const float* __restrict__ b_attn,
        unsigned short* __restrict__ Qb,
        unsigned short* __restrict__ Kb,
        unsigned short* __restrict__ Vb) {
    __shared__ unsigned short As[128 * 32];
    __shared__ unsigned short BsT[128 * 32];

    const int tid = threadIdx.x;
    const int lane = tid & 31;
    const int wv = tid >> 5;
    const int m0 = blockIdx.y * 128;
    const int n0 = blockIdx.x * 128;

    v8f acc[8] = {};
    gemm_tile_128x128<3 * D_SZ>(hs16, D_SZ, B_SZ * S_SZ, Wa16, m0, n0, tid, acc, As, BsT);

    const int half = lane >> 4;
#pragma unroll
    for (int j = 0; j < 8; ++j) {
#pragma unroll
        for (int r = 0; r < 8; ++r) {
            int n = n0 + j * 16 + (lane & 15);
            int m = m0 + wv * 16 + r + 8 * half;
            float v = acc[j][r] + b_attn[n];
            int which = n >> 10;          // 0:Q 1:K 2:V
            int c = n & 1023;
            int h = c >> 6, dd = c & 63;
            int b = m >> 11, s = m & 2047;
            unsigned short* dstbuf = (which == 0) ? Qb : ((which == 1) ? Kb : Vb);
            dstbuf[(((b << 4) + h) * S_SZ + s) * HD_SZ + dd] = f2bf(v);
        }
    }
}

// ---------------------------------------------------------------------------
// Kernel 3: flash attention per (b,h,64-row q chunk). 128 threads (4 waves),
// each wave owns a 16-row q tile; block steps 32 keys at a time.
// K tile staged via TDM; V tile staged transposed for B-fragment loads.
// ---------------------------------------------------------------------------
__global__ __launch_bounds__(128) void attention_kernel(
        const unsigned short* __restrict__ Qb,
        const unsigned short* __restrict__ Kb,
        const unsigned short* __restrict__ Vb,
        const float* __restrict__ mask,
        unsigned short* __restrict__ O16) {
    __shared__ unsigned short Ks[32 * 64];     // [key][feat]  == [N][K] for scores
    __shared__ unsigned short VsT[64 * 32];    // [feat][key]  == [N][K] for P@V
    __shared__ unsigned short Ps[4][16 * 32];  // per-wave P tile, row-major [q][key]

    const int tid = threadIdx.x;
    const int lane = tid & 31;
    const int wv = tid >> 5;
    const int half = lane >> 4;

    const int qc = blockIdx.x & 31;            // 32 q-chunks of 64 rows
    const int bh = blockIdx.x >> 5;            // b*16 + h
    const int b = bh >> 4, h = bh & 15;

    const unsigned short* Qh = Qb + (size_t)bh * S_SZ * HD_SZ;
    const unsigned short* Kh = Kb + (size_t)bh * S_SZ * HD_SZ;
    const unsigned short* Vh = Vb + (size_t)bh * S_SZ * HD_SZ;

    const int q0 = qc * 64 + wv * 16;
    v16bf qf0 = load_a_frag(Qh, HD_SZ, q0, 0, lane);
    v16bf qf1 = load_a_frag(Qh, HD_SZ, q0, 32, lane);

    float m_run[8], l_run[8];
#pragma unroll
    for (int r = 0; r < 8; ++r) { m_run[r] = -1e30f; l_run[r] = 0.0f; }
    v8f oacc[4] = {};

    for (int kb = 0; kb < S_SZ; kb += 32) {
        __syncthreads();
#ifdef HAVE_TDM
        if (wv == 0) {
            tdm_load_2d_bf16((unsigned int)(size_t)(void*)Ks,
                             (const void*)(Kh + (size_t)kb * HD_SZ),
                             HD_SZ, S_SZ, HD_SZ, 64u, 32u);
        }
#else
        {
            unsigned int* dst = (unsigned int*)Ks;
            const unsigned int* src = (const unsigned int*)Kh + kb * (HD_SZ / 2);
#pragma unroll
            for (int rep = 0; rep < 8; ++rep) dst[tid + 128 * rep] = src[tid + 128 * rep];
        }
#endif
#pragma unroll
        for (int rep = 0; rep < 16; ++rep) {
            int i = tid + 128 * rep;              // over [32][64]
            int row = i >> 6, col = i & 63;
            VsT[col * 32 + row] = Vh[(kb + row) * HD_SZ + col];
        }
#ifdef HAVE_TDM
        if (wv == 0) __builtin_amdgcn_s_wait_tensorcnt(0);
#endif
        __syncthreads();

        // Scores S = Q K^T : two 16x16 key tiles, hd=64 = 2 wmma K-steps each.
        v8f sf[2];
#pragma unroll
        for (int t = 0; t < 2; ++t) {
            v8f z = {};
            v16bf kf0 = load_b_fragT(Ks, 64, t * 16, 0, lane);
            z = __builtin_amdgcn_wmma_f32_16x16x32_bf16(false, qf0, false, kf0,
                                                        (short)0, z, false, false);
            v16bf kf1 = load_b_fragT(Ks, 64, t * 16, 32, lane);
            z = __builtin_amdgcn_wmma_f32_16x16x32_bf16(false, qf1, false, kf1,
                                                        (short)0, z, false, false);
            sf[t] = z;
        }

        float madd[2];
#pragma unroll
        for (int t = 0; t < 2; ++t) {
            int n = kb + t * 16 + (lane & 15);
            madd[t] = (1.0f - mask[b * S_SZ + n]) * -10000.0f;
        }

        // Online softmax. Row m = r + 8*half lives across 16 lanes of a half.
        float alpha[8], p0[8], p1[8];
#pragma unroll
        for (int r = 0; r < 8; ++r) {
            float s0 = sf[0][r] * 0.125f + madd[0];   // 1/sqrt(64)
            float s1 = sf[1][r] * 0.125f + madd[1];
            float mx = fmaxf(s0, s1);
#pragma unroll
            for (int d = 1; d < 16; d <<= 1) mx = fmaxf(mx, __shfl_xor(mx, d, 32));
            float mnew = fmaxf(m_run[r], mx);
            alpha[r] = __expf(m_run[r] - mnew);
            p0[r] = __expf(s0 - mnew);
            p1[r] = __expf(s1 - mnew);
            float sum = p0[r] + p1[r];
#pragma unroll
            for (int d = 1; d < 16; d <<= 1) sum += __shfl_xor(sum, d, 32);
            l_run[r] = l_run[r] * alpha[r] + sum;
            m_run[r] = mnew;
        }

        // C-layout -> A-layout for P via per-wave LDS round trip.
        unsigned short* P = Ps[wv];
#pragma unroll
        for (int r = 0; r < 8; ++r) {
            int m = r + 8 * half;
            P[m * 32 + (lane & 15)]      = f2bf(p0[r]);
            P[m * 32 + 16 + (lane & 15)] = f2bf(p1[r]);
        }
#pragma unroll
        for (int nt = 0; nt < 4; ++nt)
#pragma unroll
            for (int r = 0; r < 8; ++r) oacc[nt][r] *= alpha[r];

        v16bf pf = load_a_frag(P, 32, 0, 0, lane);
#pragma unroll
        for (int nt = 0; nt < 4; ++nt) {
            v16bf vf = load_b_fragT(VsT, 32, nt * 16, 0, lane);
            oacc[nt] = __builtin_amdgcn_wmma_f32_16x16x32_bf16(
                false, pf, false, vf, (short)0, oacc[nt], false, false);
        }
    }

#pragma unroll
    for (int nt = 0; nt < 4; ++nt) {
#pragma unroll
        for (int r = 0; r < 8; ++r) {
            int m = q0 + r + 8 * half;
            int feat = nt * 16 + (lane & 15);
            float val = oacc[nt][r] / l_run[r];
            O16[((size_t)b * S_SZ + m) * D_SZ + h * HD_SZ + feat] = f2bf(val);
        }
    }
}

// ---------------------------------------------------------------------------
// Kernel 4: output projection. [4096,1024]bf16 @ [1024,1024]bf16 + bias -> f32
// ---------------------------------------------------------------------------
__global__ __launch_bounds__(256) void proj_gemm_kernel(
        const unsigned short* __restrict__ A16,
        const unsigned short* __restrict__ Wp16,
        const float* __restrict__ b_proj,
        float* __restrict__ out) {
    __shared__ unsigned short As[128 * 32];
    __shared__ unsigned short BsT[128 * 32];

    const int tid = threadIdx.x;
    const int lane = tid & 31;
    const int wv = tid >> 5;
    const int m0 = blockIdx.y * 128;
    const int n0 = blockIdx.x * 128;

    v8f acc[8] = {};
    gemm_tile_128x128<D_SZ>(A16, D_SZ, B_SZ * S_SZ, Wp16, m0, n0, tid, acc, As, BsT);

    const int half = lane >> 4;
#pragma unroll
    for (int j = 0; j < 8; ++j) {
#pragma unroll
        for (int r = 0; r < 8; ++r) {
            int n = n0 + j * 16 + (lane & 15);
            int m = m0 + wv * 16 + r + 8 * half;
            out[(size_t)m * D_SZ + n] = acc[j][r] + b_proj[n];
        }
    }
}

// ---------------------------------------------------------------------------
// Host-side launcher
// ---------------------------------------------------------------------------
extern "C" void kernel_launch(void* const* d_in, const int* in_sizes, int n_in,
                              void* d_out, int out_size, void* d_ws, size_t ws_size,
                              hipStream_t stream) {
    (void)in_sizes; (void)n_in; (void)out_size; (void)ws_size;

    const float* hidden = (const float*)d_in[0];  // [2,2048,1024]
    const float* mask   = (const float*)d_in[1];  // [2,2048]
    const float* W_attn = (const float*)d_in[2];  // [1024,3072]
    const float* b_attn = (const float*)d_in[3];  // [3072]
    const float* W_proj = (const float*)d_in[4];  // [1024,1024]
    const float* b_proj = (const float*)d_in[5];  // [1024]
    float* out = (float*)d_out;                   // [2,2048,1024]

    const int HS_N = B_SZ * S_SZ * D_SZ;               // 4194304
    const int WA_N = D_SZ * 3 * D_SZ;                  // 3145728
    const int WP_N = D_SZ * D_SZ;                      // 1048576
    const int HEADBUF_N = B_SZ * H_SZ * S_SZ * HD_SZ;  // 4194304

    char* ws = (char*)d_ws;
    size_t off = 0;
    unsigned short* hs16 = (unsigned short*)(ws + off); off += (size_t)HS_N * 2;
    unsigned short* Wa16 = (unsigned short*)(ws + off); off += (size_t)WA_N * 2;
    unsigned short* Wp16 = (unsigned short*)(ws + off); off += (size_t)WP_N * 2;
    unsigned short* Qb   = (unsigned short*)(ws + off); off += (size_t)HEADBUF_N * 2;
    unsigned short* Kb   = (unsigned short*)(ws + off); off += (size_t)HEADBUF_N * 2;
    unsigned short* Vb   = (unsigned short*)(ws + off); off += (size_t)HEADBUF_N * 2;
    unsigned short* O16  = (unsigned short*)(ws + off); off += (size_t)HS_N * 2;

    cvt_bf16_kernel<<<(HS_N + 255) / 256, 256, 0, stream>>>(hidden, hs16, HS_N);
    cvt_bf16_kernel<<<(WA_N + 255) / 256, 256, 0, stream>>>(W_attn, Wa16, WA_N);
    cvt_bf16_kernel<<<(WP_N + 255) / 256, 256, 0, stream>>>(W_proj, Wp16, WP_N);

    // QKV GEMM: grid (3072/128, 4096/128)
    qkv_gemm_kernel<<<dim3(24, 32), 256, 0, stream>>>(hs16, Wa16, b_attn, Qb, Kb, Vb);

    // Flash attention: B*H*(S/64) = 1024 blocks, 128 threads
    attention_kernel<<<dim3(B_SZ * H_SZ * (S_SZ / 64)), 128, 0, stream>>>(
        Qb, Kb, Vb, mask, O16);

    // Output projection: grid (1024/128, 4096/128)
    proj_gemm_kernel<<<dim3(8, 32), 256, 0, stream>>>(O16, Wp16, b_proj, out);
}
